// GPT2nt_33724083208555
// MI455X (gfx1250) — compile-verified, matrix operand order checked by reading, not run on hardware
//
#include <hip/hip_runtime.h>
#include <hip/hip_bf16.h>
#include <math.h>

// ---------------------------------------------------------------------------
// GPT-2 style 6-layer transformer forward for MI455X (gfx1250).
// - all GEMMs + attention on v_wmma_f32_16x16x32_bf16 (wave32 WMMA)
// - weights pre-converted fp32->bf16 once (halves L2 panel traffic)
// - GEMM tiles staged with async global->LDS copies (ASYNCcnt path)
// ---------------------------------------------------------------------------

typedef __attribute__((ext_vector_type(16))) __bf16 v16bf;
typedef __attribute__((ext_vector_type(8)))  __bf16 v8bf;
typedef __attribute__((ext_vector_type(8)))  float  v8f;

#define N_EMBD 768
#define N_HEAD 12
#define HD     64
#define TT     2560            // total sequence length (10 segments x 256)
#define BB     2
#define MROWS  (BB*TT)         // 5120
#define CMID   (4*N_EMBD)      // 3072
#define NLAYER 6

enum { EP_F32_RES = 0, EP_BF16_GELU = 1, EP_BF16_HEADS = 2 };

static __device__ __forceinline__ __bf16 f2bf(float f) {
  union { float f; unsigned u; } x; x.f = f;
  unsigned r = x.u + 0x7FFFu + ((x.u >> 16) & 1u);   // round-to-nearest-even
  unsigned short h = (unsigned short)(r >> 16);
  union { unsigned short s; __bf16 b; } y; y.s = h; return y.b;
}

// Assemble a 16-bit A/B WMMA fragment: VGPRs 0-3 hold K=kb..kb+7,
// VGPRs 4-7 hold K=kb+16..kb+23 (per CDNA5 ISA 16-bit 16x32 layout).
static __device__ __forceinline__ v16bf mk_frag(v8bf lo, v8bf hi) {
  v16bf r;
#pragma unroll
  for (int i = 0; i < 8; ++i) { r[i] = lo[i]; r[i + 8] = hi[i]; }
  return r;
}

static __device__ __forceinline__ v16bf lds_frag(const __bf16* rowptr, int kb) {
  v8bf lo = *(const v8bf*)(rowptr + kb);
  v8bf hi = *(const v8bf*)(rowptr + kb + 16);
  return mk_frag(lo, hi);
}

// Async global->LDS 16-byte copy (GLOBAL_LOAD_ASYNC_TO_LDS_B128, ASYNCcnt).
static __device__ __forceinline__ void async_cp16(const __bf16* lds_dst,
                                                  const __bf16* gsrc) {
  unsigned lds_off = (unsigned)(size_t)lds_dst;       // low 32b = LDS offset
  unsigned long long ga = (unsigned long long)(size_t)gsrc;
  asm volatile("global_load_async_to_lds_b128 %0, %1, off"
               :: "v"(lds_off), "v"(ga) : "memory");
}
static __device__ __forceinline__ void wait_async0() {
  asm volatile("s_wait_asynccnt 0x0" ::: "memory");
}

// ---------------------------------------------------------------------------
// fp32 -> bf16 bulk conversion (weights), 8 elements / thread.
// ---------------------------------------------------------------------------
__global__ __launch_bounds__(256) void convw_k(const float* __restrict__ src,
                                               __bf16* __restrict__ dst,
                                               long long n) {
  long long i = ((long long)blockIdx.x * 256 + threadIdx.x) * 8;
  if (i >= n) return;
  float4 a = *(const float4*)(src + i);
  float4 b = *(const float4*)(src + i + 4);
  v8bf o;
  o[0] = f2bf(a.x); o[1] = f2bf(a.y); o[2] = f2bf(a.z); o[3] = f2bf(a.w);
  o[4] = f2bf(b.x); o[5] = f2bf(b.y); o[6] = f2bf(b.z); o[7] = f2bf(b.w);
  *(v8bf*)(dst + i) = o;
}

// ---------------------------------------------------------------------------
// x = concat(5 cond, 5 idx) + pos_emb
// ---------------------------------------------------------------------------
__global__ __launch_bounds__(256) void embed_k(
    const float* __restrict__ iu,  const float* __restrict__ idn,
    const float* __restrict__ il,  const float* __restrict__ ir,
    const float* __restrict__ itr, const float* __restrict__ cm,
    const float* __restrict__ cu,  const float* __restrict__ cd,
    const float* __restrict__ cl,  const float* __restrict__ cr,
    const float* __restrict__ pos, float* __restrict__ x) {
  int idx = blockIdx.x * 256 + threadIdx.x;
  const int TC = TT * N_EMBD;
  if (idx >= BB * TC) return;
  int b = idx / TC, rem = idx - b * TC;
  int t = rem / N_EMBD, c = rem - t * N_EMBD;
  int seg = t >> 8, tt = t & 255;
  const float* srcs[10] = { cm, cu, cd, cl, cr, iu, idn, il, ir, itr };
  float tok = srcs[seg][((size_t)b * 256 + tt) * N_EMBD + c];
  x[idx] = tok + pos[(size_t)t * N_EMBD + c];
}

// ---------------------------------------------------------------------------
// LayerNorm over C=768, bf16 output (GEMM input). One block per row.
// ---------------------------------------------------------------------------
__global__ __launch_bounds__(256) void ln_bf16(
    const float* __restrict__ x, const float* __restrict__ w,
    const float* __restrict__ b, __bf16* __restrict__ out) {
  __shared__ float s1[256], s2[256];
  int row = blockIdx.x, tid = threadIdx.x;
  const float* xr = x + (size_t)row * N_EMBD;
  float v0 = xr[tid], v1 = xr[tid + 256], v2 = xr[tid + 512];
  s1[tid] = v0 + v1 + v2;
  s2[tid] = v0 * v0 + v1 * v1 + v2 * v2;
  __syncthreads();
  for (int o = 128; o > 0; o >>= 1) {
    if (tid < o) { s1[tid] += s1[tid + o]; s2[tid] += s2[tid + o]; }
    __syncthreads();
  }
  float mu  = s1[0] * (1.0f / 768.0f);
  float var = s2[0] * (1.0f / 768.0f) - mu * mu;
  float rs  = rsqrtf(var + 1e-5f);
  __bf16* orow = out + (size_t)row * N_EMBD;
  orow[tid]       = f2bf((v0 - mu) * rs * w[tid]       + b[tid]);
  orow[tid + 256] = f2bf((v1 - mu) * rs * w[tid + 256] + b[tid + 256]);
  orow[tid + 512] = f2bf((v2 - mu) * rs * w[tid + 512] + b[tid + 512]);
}

// ---------------------------------------------------------------------------
// out[M,N] = epilogue( A_bf16[M,K] @ Wb_bf16[N,K]^T + bias[N] )
// 128x64 block tile, 256 threads = 8 waves (4m x 2n), each wave 32x32:
// 4 WMMA per K=32 step.  Tiles staged via async global->LDS b128 copies.
// ---------------------------------------------------------------------------
__global__ __launch_bounds__(256) void gemm_bf16_wmma(
    const __bf16* __restrict__ A, const __bf16* __restrict__ Wb,
    const float* __restrict__ bias, const float* __restrict__ resid,
    float* __restrict__ outF, __bf16* __restrict__ outB,
    int M, int N, int K, int mode) {
  __shared__ __bf16 As[128 * 40];  // stride 40 halfs: conflict-free b128
  __shared__ __bf16 Bs[64 * 40];

  const int tid = threadIdx.x;
  const int bm = blockIdx.y * 128, bn = blockIdx.x * 64;
  const int ar = tid >> 1, ac = (tid & 1) * 16;    // A loader: 128 rows x 32
  const int br = tid >> 2, bc = (tid & 3) * 8;     // B loader: 64 rows x 32
  const int ln = tid & 31, wv = tid >> 5;
  const int l16 = ln & 15, lh = ln >> 4, kb = lh * 8;
  const int mw = (wv & 3) * 32, nw = (wv >> 2) * 32;

  v8f c00 = {}, c01 = {}, c10 = {}, c11 = {};
  const __bf16* ag = A  + (size_t)(bm + ar) * K + ac;
  const __bf16* wg = Wb + (size_t)(bn + br) * K + bc;

  for (int k0 = 0; k0 < K; k0 += 32) {
    async_cp16(&As[ar * 40 + ac],     ag + k0);
    async_cp16(&As[ar * 40 + ac + 8], ag + k0 + 8);
    async_cp16(&Bs[br * 40 + bc],     wg + k0);
    if (k0 + 32 < K) {                 // pull next K tile toward near caches
      __builtin_prefetch(ag + k0 + 32, 0, 3);
      __builtin_prefetch(wg + k0 + 32, 0, 3);
    }
    wait_async0();
    __syncthreads();
    v16bf a0 = lds_frag(&As[(mw + l16) * 40], kb);
    v16bf a1 = lds_frag(&As[(mw + 16 + l16) * 40], kb);
    v16bf b0 = lds_frag(&Bs[(nw + l16) * 40], kb);
    v16bf b1 = lds_frag(&Bs[(nw + 16 + l16) * 40], kb);
    c00 = __builtin_amdgcn_wmma_f32_16x16x32_bf16(false, a0, false, b0, (short)0, c00, false, false);
    c01 = __builtin_amdgcn_wmma_f32_16x16x32_bf16(false, a0, false, b1, (short)0, c01, false, false);
    c10 = __builtin_amdgcn_wmma_f32_16x16x32_bf16(false, a1, false, b0, (short)0, c10, false, false);
    c11 = __builtin_amdgcn_wmma_f32_16x16x32_bf16(false, a1, false, b1, (short)0, c11, false, false);
    __syncthreads();
  }

  // Epilogue.  C/D layout: element (m,n): m = r + 8*(lane/16), n = lane%16.
#pragma unroll
  for (int im = 0; im < 2; ++im) {
#pragma unroll
    for (int in = 0; in < 2; ++in) {
      v8f cc = im ? (in ? c11 : c10) : (in ? c01 : c00);
      int ncol = bn + nw + in * 16 + l16;
      float bv = bias[ncol];
#pragma unroll
      for (int r = 0; r < 8; ++r) {
        int row = bm + mw + im * 16 + r + 8 * lh;
        float vv = cc[r] + bv;
        if (mode == EP_F32_RES) {
          outF[(size_t)row * N + ncol] = vv + resid[(size_t)row * N + ncol];
        } else if (mode == EP_BF16_GELU) {
          float g = 0.5f * vv * (1.0f + erff(vv * 0.70710678f));
          outB[(size_t)row * N + ncol] = f2bf(g);
        } else {  // EP_BF16_HEADS: [B*T, C] -> [B, H, T, hd] bf16
          int bidx = row / TT, t = row - bidx * TT;
          int h = ncol >> 6, d = ncol & 63;
          outB[(((size_t)bidx * N_HEAD + h) * TT + t) * HD + d] = f2bf(vv);
        }
      }
    }
  }
}

// ---------------------------------------------------------------------------
// Flash-style attention.  Block = 4 waves = 64 q rows for one (b,h).
// Mask: valid iff (k % 256) <= (q % 256)  (tiled block-causal).
// ---------------------------------------------------------------------------
__global__ __launch_bounds__(128) void attn_wmma(
    const __bf16* __restrict__ q, const __bf16* __restrict__ k,
    const __bf16* __restrict__ v, __bf16* __restrict__ y) {
  __shared__ __bf16 ktile[32 * 72];      // [key][d], padded stride 72
  __shared__ __bf16 vtile[64 * 40];      // [d][key] (transposed), stride 40
  __shared__ __bf16 Pb[4 * 16 * 40];     // per-wave P staging

  const int tid = threadIdx.x;
  const int wv = tid >> 5, ln = tid & 31, l16 = ln & 15, lh = ln >> 4, kbo = lh * 8;
  const int qblk = blockIdx.x, bh = blockIdx.y;
  const int b = bh / N_HEAD, h = bh - b * N_HEAD;
  const int qtile = qblk * 64 + wv * 16;
  const size_t base = (size_t)bh * TT * HD;

  const __bf16* qrow = q + base + (size_t)(qtile + l16) * HD;
  v16bf qa0 = mk_frag(*(const v8bf*)(qrow + kbo), *(const v8bf*)(qrow + kbo + 16));
  v16bf qa1 = mk_frag(*(const v8bf*)(qrow + 32 + kbo), *(const v8bf*)(qrow + 32 + kbo + 16));

  float mr[8], lr[8], alpha[8];
  v8f oc0 = {}, oc1 = {}, oc2 = {}, oc3 = {};
#pragma unroll
  for (int r = 0; r < 8; ++r) { mr[r] = -1e30f; lr[r] = 0.f; }

  const int qmodB = (qblk * 64) & 255;
  const int qmodW = (qblk * 64 + wv * 16) & 255;
  __bf16* Pw = Pb + wv * 16 * 40;

  for (int kt0 = 0; kt0 < TT; kt0 += 32) {
    int km0 = kt0 & 255;                 // key tile never crosses a segment
    if (km0 > qmodB + 63) continue;      // fully masked for whole block

    // Stage K [32,64] and V^T [64,32] into LDS (128 threads x 2 chunks).
    {
      int h0 = tid * 8;
#pragma unroll
      for (int cch = 0; cch < 2; ++cch) {
        int hh = h0 + cch * 1024;
        int key = hh >> 6, d0 = hh & 63;
        *(v8bf*)&ktile[key * 72 + d0] =
            *(const v8bf*)(k + base + (size_t)(kt0 + key) * HD + d0);
        v8bf vv = *(const v8bf*)(v + base + (size_t)(kt0 + key) * HD + d0);
#pragma unroll
        for (int i = 0; i < 8; ++i) vtile[(d0 + i) * 40 + key] = vv[i];
      }
    }
    __syncthreads();

    // S = q . k^T  (16 q rows x 32 keys), K-reduction over hd=64.
    v8f s0 = {}, s1 = {};
    {
      v16bf kf0 = lds_frag(&ktile[(l16) * 72], kbo);
      v16bf kf1 = lds_frag(&ktile[(16 + l16) * 72], kbo);
      s0 = __builtin_amdgcn_wmma_f32_16x16x32_bf16(false, qa0, false, kf0, (short)0, s0, false, false);
      s1 = __builtin_amdgcn_wmma_f32_16x16x32_bf16(false, qa0, false, kf1, (short)0, s1, false, false);
      v16bf kf2 = lds_frag(&ktile[(l16) * 72], 32 + kbo);
      v16bf kf3 = lds_frag(&ktile[(16 + l16) * 72], 32 + kbo);
      s0 = __builtin_amdgcn_wmma_f32_16x16x32_bf16(false, qa1, false, kf2, (short)0, s0, false, false);
      s1 = __builtin_amdgcn_wmma_f32_16x16x32_bf16(false, qa1, false, kf3, (short)0, s1, false, false);
    }

    // Scale + mask.
#pragma unroll
    for (int r = 0; r < 8; ++r) {
      int qm = qmodW + r + 8 * lh;
      float xa = s0[r] * 0.125f; if (km0 + l16 > qm)      xa = -1e30f;
      float xb = s1[r] * 0.125f; if (km0 + 16 + l16 > qm) xb = -1e30f;
      s0[r] = xa; s1[r] = xb;
    }

    // Online softmax (row reductions across the 16 lanes of each half).
#pragma unroll
    for (int r = 0; r < 8; ++r) {
      float t_ = fmaxf(s0[r], s1[r]);
      t_ = fmaxf(t_, __shfl_xor(t_, 1));
      t_ = fmaxf(t_, __shfl_xor(t_, 2));
      t_ = fmaxf(t_, __shfl_xor(t_, 4));
      t_ = fmaxf(t_, __shfl_xor(t_, 8));
      float nm = fmaxf(mr[r], t_);
      alpha[r] = __expf(mr[r] - nm);
      mr[r] = nm;
      float pa = __expf(s0[r] - nm), pb = __expf(s1[r] - nm);
      s0[r] = pa; s1[r] = pb;
      float ps = pa + pb;
      ps += __shfl_xor(ps, 1); ps += __shfl_xor(ps, 2);
      ps += __shfl_xor(ps, 4); ps += __shfl_xor(ps, 8);
      lr[r] = lr[r] * alpha[r] + ps;
    }
#pragma unroll
    for (int r = 0; r < 8; ++r) {
      oc0[r] *= alpha[r]; oc1[r] *= alpha[r];
      oc2[r] *= alpha[r]; oc3[r] *= alpha[r];
    }

    // Re-layout P (C-layout -> A-fragment) through per-wave LDS.
#pragma unroll
    for (int r = 0; r < 8; ++r) {
      int rp = r + 8 * lh;
      Pw[rp * 40 + l16]      = f2bf(s0[r]);
      Pw[rp * 40 + 16 + l16] = f2bf(s1[r]);
    }
    v16bf pa  = lds_frag(&Pw[l16 * 40], kbo);         // same-wave LDS: in-order
    v16bf vf0 = lds_frag(&vtile[(l16) * 40], kbo);
    v16bf vf1 = lds_frag(&vtile[(16 + l16) * 40], kbo);
    v16bf vf2 = lds_frag(&vtile[(32 + l16) * 40], kbo);
    v16bf vf3 = lds_frag(&vtile[(48 + l16) * 40], kbo);
    oc0 = __builtin_amdgcn_wmma_f32_16x16x32_bf16(false, pa, false, vf0, (short)0, oc0, false, false);
    oc1 = __builtin_amdgcn_wmma_f32_16x16x32_bf16(false, pa, false, vf1, (short)0, oc1, false, false);
    oc2 = __builtin_amdgcn_wmma_f32_16x16x32_bf16(false, pa, false, vf2, (short)0, oc2, false, false);
    oc3 = __builtin_amdgcn_wmma_f32_16x16x32_bf16(false, pa, false, vf3, (short)0, oc3, false, false);
    __syncthreads();
  }

  // Normalize and write y as bf16 in [B, T, C] layout for the proj GEMM.
#pragma unroll
  for (int r = 0; r < 8; ++r) {
    float inv = 1.0f / lr[r];
    int trow = qtile + r + 8 * lh;
    size_t orow = ((size_t)b * TT + trow) * N_EMBD + (size_t)h * HD;
    y[orow + l16]      = f2bf(oc0[r] * inv);
    y[orow + 16 + l16] = f2bf(oc1[r] * inv);
    y[orow + 32 + l16] = f2bf(oc2[r] * inv);
    y[orow + 48 + l16] = f2bf(oc3[r] * inv);
  }
}

// ---------------------------------------------------------------------------
extern "C" void kernel_launch(void* const* d_in, const int* in_sizes, int n_in,
                              void* d_out, int out_size, void* d_ws, size_t ws_size,
                              hipStream_t stream) {
  (void)in_sizes; (void)n_in; (void)out_size; (void)ws_size;
  const float* idx_up     = (const float*)d_in[0];
  const float* idx_down   = (const float*)d_in[1];
  const float* idx_lhand  = (const float*)d_in[2];
  const float* idx_rhand  = (const float*)d_in[3];
  const float* idx_transl = (const float*)d_in[4];
  const float* cond_music = (const float*)d_in[5];
  const float* cond_up    = (const float*)d_in[6];
  const float* cond_down  = (const float*)d_in[7];
  const float* cond_lhand = (const float*)d_in[8];
  const float* cond_rhand = (const float*)d_in[9];
  const float* pos_emb = (const float*)d_in[10];
  const float* ln1_w = (const float*)d_in[11];
  const float* ln1_b = (const float*)d_in[12];
  const float* ln2_w = (const float*)d_in[13];
  const float* ln2_b = (const float*)d_in[14];
  const float* Wq = (const float*)d_in[15];
  const float* Wk = (const float*)d_in[16];
  const float* Wv = (const float*)d_in[17];
  const float* Wp = (const float*)d_in[18];
  const float* bq = (const float*)d_in[19];
  const float* bk = (const float*)d_in[20];
  const float* bv = (const float*)d_in[21];
  const float* bp = (const float*)d_in[22];
  const float* W1 = (const float*)d_in[23];
  const float* b1 = (const float*)d_in[24];
  const float* W2 = (const float*)d_in[25];
  const float* b2 = (const float*)d_in[26];

  char* ws = (char*)d_ws;
  size_t off = 0;
  auto alloc = [&](size_t bytes) {
    void* p = ws + off; off += (bytes + 255) & ~(size_t)255; return p;
  };
  float*  x    = (float*) alloc((size_t)MROWS * N_EMBD * 4);
  __bf16* hb   = (__bf16*)alloc((size_t)MROWS * N_EMBD * 2);
  __bf16* qbuf = (__bf16*)alloc((size_t)MROWS * N_EMBD * 2);
  __bf16* kbuf = (__bf16*)alloc((size_t)MROWS * N_EMBD * 2);
  __bf16* vbuf = (__bf16*)alloc((size_t)MROWS * N_EMBD * 2);
  __bf16* ybuf = (__bf16*)alloc((size_t)MROWS * N_EMBD * 2);
  __bf16* gbuf = (__bf16*)alloc((size_t)MROWS * CMID * 2);
  const long long nCC = (long long)NLAYER * N_EMBD * N_EMBD;   // per QKV/P set
  const long long nMC = (long long)NLAYER * CMID * N_EMBD;
  __bf16* Wqb = (__bf16*)alloc((size_t)nCC * 2);
  __bf16* Wkb = (__bf16*)alloc((size_t)nCC * 2);
  __bf16* Wvb = (__bf16*)alloc((size_t)nCC * 2);
  __bf16* Wpb = (__bf16*)alloc((size_t)nCC * 2);
  __bf16* W1b = (__bf16*)alloc((size_t)nMC * 2);
  __bf16* W2b = (__bf16*)alloc((size_t)nMC * 2);

  // One-shot weight downconversion fp32 -> bf16.
  {
    int gcc = (int)((nCC + 2047) / 2048), gmc = (int)((nMC + 2047) / 2048);
    convw_k<<<gcc, 256, 0, stream>>>(Wq, Wqb, nCC);
    convw_k<<<gcc, 256, 0, stream>>>(Wk, Wkb, nCC);
    convw_k<<<gcc, 256, 0, stream>>>(Wv, Wvb, nCC);
    convw_k<<<gcc, 256, 0, stream>>>(Wp, Wpb, nCC);
    convw_k<<<gmc, 256, 0, stream>>>(W1, W1b, nMC);
    convw_k<<<gmc, 256, 0, stream>>>(W2, W2b, nMC);
  }

  embed_k<<<(MROWS * N_EMBD + 255) / 256, 256, 0, stream>>>(
      idx_up, idx_down, idx_lhand, idx_rhand, idx_transl,
      cond_music, cond_up, cond_down, cond_lhand, cond_rhand, pos_emb, x);

  const dim3 gC(N_EMBD / 64, MROWS / 128);   // N=768 GEMMs
  const dim3 gM(CMID / 64,  MROWS / 128);    // N=3072 GEMM
  const dim3 gA(TT / 64, BB * N_HEAD);       // attention

  for (int li = 0; li < NLAYER; ++li) {
    const size_t wCC = (size_t)li * N_EMBD * N_EMBD;
    const size_t wMC = (size_t)li * CMID * N_EMBD;

    ln_bf16<<<MROWS, 256, 0, stream>>>(x, ln1_w + li * N_EMBD, ln1_b + li * N_EMBD, hb);

    gemm_bf16_wmma<<<gC, 256, 0, stream>>>(hb, Wqb + wCC, bq + li * N_EMBD, nullptr,
        nullptr, qbuf, MROWS, N_EMBD, N_EMBD, EP_BF16_HEADS);
    gemm_bf16_wmma<<<gC, 256, 0, stream>>>(hb, Wkb + wCC, bk + li * N_EMBD, nullptr,
        nullptr, kbuf, MROWS, N_EMBD, N_EMBD, EP_BF16_HEADS);
    gemm_bf16_wmma<<<gC, 256, 0, stream>>>(hb, Wvb + wCC, bv + li * N_EMBD, nullptr,
        nullptr, vbuf, MROWS, N_EMBD, N_EMBD, EP_BF16_HEADS);

    attn_wmma<<<gA, 128, 0, stream>>>(qbuf, kbuf, vbuf, ybuf);

    gemm_bf16_wmma<<<gC, 256, 0, stream>>>(ybuf, Wpb + wCC, bp + li * N_EMBD, x,
        x, nullptr, MROWS, N_EMBD, N_EMBD, EP_F32_RES);

    ln_bf16<<<MROWS, 256, 0, stream>>>(x, ln2_w + li * N_EMBD, ln2_b + li * N_EMBD, hb);

    gemm_bf16_wmma<<<gM, 256, 0, stream>>>(hb, W1b + wMC, b1 + li * CMID, nullptr,
        nullptr, gbuf, MROWS, CMID, N_EMBD, EP_BF16_GELU);

    float* outx = (li == NLAYER - 1) ? (float*)d_out : x;
    gemm_bf16_wmma<<<gC, 256, 0, stream>>>(gbuf, W2b + wMC, b2 + li * N_EMBD, x,
        outx, nullptr, MROWS, N_EMBD, CMID, EP_F32_RES);
  }
}
